// Net_24936580120706
// MI455X (gfx1250) — matrix operand order repaired and checked by reading, not compiled, not run on hardware
//
#include <hip/hip_runtime.h>
#include <hip/hip_bf16.h>
#include <math.h>

typedef __attribute__((ext_vector_type(16))) __bf16 v16bf;
typedef __attribute__((ext_vector_type(2)))  __bf16 v2bf;
typedef __attribute__((ext_vector_type(8)))  float  v8f;
typedef __attribute__((ext_vector_type(8)))  int    v8i;
typedef __attribute__((ext_vector_type(2)))  float  v2f;

#define C_  6
#define B_  4096
#define T_  512
#define H_  26
#define IN_ 2
#define F_  (C_*H_)   // 156
#define N1_ 300
#define N2_ 50
#define N3_ 14

// ---- fast tanh: CDNA5 V_TANH_F32 trans op (confirmed lowering), else
// branch-free 1 - 2/(exp2(2*log2e*x)+1) via v_exp_f32 + v_rcp_f32.
#if defined(__has_builtin)
#if __has_builtin(__builtin_amdgcn_tanhf)
#define FAST_TANH(x) __builtin_amdgcn_tanhf((x))
#endif
#endif
#ifndef FAST_TANH
__device__ __forceinline__ float fast_tanh_impl(float x) {
  float e = __builtin_amdgcn_exp2f(x * 2.88539008177792681472f); // 2*log2(e)
  float r = __builtin_amdgcn_rcpf(e + 1.0f);
  return fmaf(-2.0f, r, 1.0f);
}
#define FAST_TANH(x) fast_tanh_impl((x))
#endif

// ---- half-wave (lane ^ 16) swap: v_permlanex16_b32 identity lane-sel
// (pure VALU, no LDS path; confirmed lowering), fallback to __shfl_xor.
#if defined(__has_builtin)
#if __has_builtin(__builtin_amdgcn_permlanex16)
#define SWAP16U(x) ((unsigned)__builtin_amdgcn_permlanex16((int)(x), (int)(x), \
                     0x76543210, 0xfedcba98, false, false))
#endif
#endif
#ifndef SWAP16U
#define SWAP16U(x) ((unsigned)__shfl_xor((int)(x), 16, 32))
#endif

// ---- two-operand v_cvt_pk_bf16_f32 (confirmed fused lowering)
__device__ __forceinline__ unsigned pk_bf16(float a, float b) {
#if defined(__has_builtin) && __has_builtin(__builtin_amdgcn_cvt_pk_bf16_f32)
  v2bf p = __builtin_amdgcn_cvt_pk_bf16_f32(a, b);
  return __builtin_bit_cast(unsigned, p);
#else
  v2f f; f.x = a; f.y = b;
  v2bf p = __builtin_convertvector(f, v2bf);
  return __builtin_bit_cast(unsigned, p);
#endif
}
__device__ __forceinline__ float lo_bf16(unsigned u) {
  union { unsigned u; __bf16 h[2]; } q; q.u = u; return (float)q.h[0];
}
__device__ __forceinline__ float hi_bf16(unsigned u) {
  union { unsigned u; __bf16 h[2]; } q; q.u = u; return (float)q.h[1];
}

// Combined (augmented) recurrent weight matrix W_aug (32x32, row m = output h,
// col k = input):  k<26 -> W_hh[c][m][k], k==26/27 -> W_ih[c][m][0/1],
// k==28 -> b_ih+b_hh, else 0.  Folds x-term and both biases into one WMMA.
__device__ __forceinline__ float wcomb(const float* __restrict__ Whh,
                                       const float* __restrict__ Wih,
                                       const float* __restrict__ bih,
                                       const float* __restrict__ bhh,
                                       int c, int m, int k) {
  if (m >= H_) return 0.0f;
  if (k < H_)      return Whh[(c*H_ + m)*H_ + k];
  if (k == H_)     return Wih[(c*H_ + m)*IN_ + 0];
  if (k == H_+1)   return Wih[(c*H_ + m)*IN_ + 1];
  if (k == H_+2)   return bih[c*H_ + m] + bhh[c*H_ + m];
  return 0.0f;
}

// One wave owns a 16-batch tile of one channel and iterates the reversed-time
// recurrence  pre^T = W_aug * [h ; x_t ; 1]^T  via two v_wmma_f32_16x16x32_bf16
// per step.  h state is carried as packed-bf16 dwords in the D/B layout; the
// only cross-lane fixup per step is 4 identity v_permlanex16 swaps.  The
// length mask and the x prefetch are both branch-free.
__global__ __launch_bounds__(64, 1)
void rnn_kernel(const float* __restrict__ x, const int* __restrict__ lengths,
                const float* __restrict__ Wih, const float* __restrict__ Whh,
                const float* __restrict__ bih, const float* __restrict__ bhh,
                float* __restrict__ feats) {
  const int lane  = threadIdx.x & 31;
  const int wave  = threadIdx.x >> 5;
  const int c     = blockIdx.x % C_;
  const int b0    = (blockIdx.x / C_) * 32 + wave * 16;
  const int col   = lane & 15;   // batch column (B/C/D layouts) and A-row m
  const int hhalf = lane >> 4;   // half-wave id

  // ---- A tiles (constant over the T loop), bf16 A-layout:
  // lane: m = col; dwords v=0..7: K = 16*hhalf + 2v
  v16bf A0, A1;
#pragma unroll
  for (int v = 0; v < 8; ++v) {
    const int k0 = 16*hhalf + 2*v;
    A0[2*v]   = (__bf16)wcomb(Whh, Wih, bih, bhh, c, col,      k0);
    A0[2*v+1] = (__bf16)wcomb(Whh, Wih, bih, bhh, c, col,      k0+1);
    A1[2*v]   = (__bf16)wcomb(Whh, Wih, bih, bhh, c, 16 + col, k0);
    A1[2*v+1] = (__bf16)wcomb(Whh, Wih, bih, bhh, c, 16 + col, k0+1);
  }

  const int lenv = lengths[(size_t)(b0 + col)*C_ + c];     // mask: step < len
  // wave-max length: steps beyond it are guaranteed no-ops -> early exit
  int ml = lenv;
#pragma unroll
  for (int ofs = 1; ofs <= 8; ofs <<= 1) {
    int o = __shfl_xor(ml, ofs, 32);
    ml = (o > ml) ? o : ml;
  }

  const float* xrow = x + ((size_t)c*B_ + (b0 + col)) * T_ * IN_;

  // Masked h state, packed bf16 pairs, D layout: pA = rows 0..15, pB = 16..25
  unsigned pA[4] = {0u, 0u, 0u, 0u};
  unsigned pB[4] = {0u, 0u, 0u, 0u};
  const unsigned onebf = 0x00003F80u;   // {bf16 1.0, bf16 0.0}

  v2f xc = __builtin_nontemporal_load((const v2f*)(xrow + (size_t)(T_-1)*IN_));
  for (int step = 0; step < ml; ++step) {
    // unconditional software prefetch of x for the next step: clamp the time
    // index instead of guarding (any t in [0,T-1] is in-bounds -> no branch)
    int tn = T_ - 2 - step;  tn = (tn < 0) ? 0 : tn;      // scalar s_max
    v2f xn = __builtin_nontemporal_load((const v2f*)(xrow + (size_t)tn*IN_));

    // half-wave exchange: each lane contributes the tile its partner needs
    unsigned sw[4];
#pragma unroll
    for (int i = 0; i < 4; ++i) {
      unsigned cn = hhalf ? pA[i] : pB[i];
      sw[i] = SWAP16U(cn);
    }

    // Assemble B operand (h^T augmented with x and 1), bf16 B-layout:
    // lane: n = col, K = 16*hhalf + {2v,2v+1}
    const unsigned pkx = pk_bf16(xc.x, xc.y);
    v8i bi;
    bi[0] = (int)(hhalf ? sw[0] : pA[0]);
    bi[1] = (int)(hhalf ? sw[1] : pA[1]);
    bi[2] = (int)(hhalf ? sw[2] : pA[2]);
    bi[3] = (int)(hhalf ? sw[3] : pA[3]);
    bi[4] = (int)(hhalf ? pB[0] : sw[0]);   // K=24,25 (h rows 24,25)
    bi[5] = (int)(hhalf ? pkx   : sw[1]);   // K=26,27 -> x0, x1
    bi[6] = (int)(hhalf ? onebf : sw[2]);   // K=28,29 -> 1, 0
    bi[7] = (int)(hhalf ? 0u    : sw[3]);   // K=30,31 -> 0, 0
    v16bf Bv = __builtin_bit_cast(v16bf, bi);

    const v8f z = {};
    v8f d0 = __builtin_amdgcn_wmma_f32_16x16x32_bf16(false, A0, false, Bv,
                                                     (short)0, z, false, false);
    v8f d1 = __builtin_amdgcn_wmma_f32_16x16x32_bf16(false, A1, false, Bv,
                                                     (short)0, z, false, false);

    // tanh (hw trans op) + length mask as a pure data-flow select
    const unsigned sel = (step < lenv) ? 0xFFFFFFFFu : 0u;
#pragma unroll
    for (int i = 0; i < 4; ++i) {
      unsigned nA = pk_bf16(FAST_TANH(d0[2*i]), FAST_TANH(d0[2*i+1]));
      unsigned nB = pk_bf16(FAST_TANH(d1[2*i]), FAST_TANH(d1[2*i+1]));
      pA[i] = (nA & sel) | (pA[i] & ~sel);
      pB[i] = (nB & sel) | (pB[i] & ~sel);
    }
    xc = xn;
  }

  // feats[b][c*26 + (25 - hidx)] = relu(h)   (the hs[:,:,::-1] + relu(f) fold)
  float* dst = feats + (size_t)(b0 + col)*F_ + c*H_;
#pragma unroll
  for (int i = 0; i < 4; ++i) {
    const int r0 = 2*i, r1 = 2*i + 1;
    const int hA0 = r0 + 8*hhalf,      hA1 = r1 + 8*hhalf;       // 0..15
    const int hB0 = 16 + r0 + 8*hhalf, hB1 = 16 + r1 + 8*hhalf;  // 16..31
    dst[H_-1 - hA0] = fmaxf(lo_bf16(pA[i]), 0.0f);
    dst[H_-1 - hA1] = fmaxf(hi_bf16(pA[i]), 0.0f);
    if (hB0 < H_) dst[H_-1 - hB0] = fmaxf(lo_bf16(pB[i]), 0.0f);
    if (hB1 < H_) dst[H_-1 - hB1] = fmaxf(hi_bf16(pB[i]), 0.0f);
  }
}

// Layer 1: h1 = relu(feats @ W1^T + b1), M=4096 N=300 K=156 -> bf16 WMMA,
// one 16x16 tile per wave, 5 K-chunks of 32 (padded to 160).
__global__ __launch_bounds__(256, 1)
void mlp1_kernel(const float* __restrict__ feats, const float* __restrict__ W1,
                 const float* __restrict__ b1, float* __restrict__ h1) {
  const int lane  = threadIdx.x & 31;
  const int wave  = threadIdx.x >> 5;
  const int wid   = blockIdx.x * 8 + wave;
  const int NT    = 19;                 // ceil(300/16)
  const int mtile = wid / NT, ntile = wid % NT;
  const int m0 = mtile * 16, n0 = ntile * 16;
  const int col = lane & 15, hhalf = lane >> 4;

  v8f acc = {};
  for (int kc = 0; kc < 5; ++kc) {
    const int kbase = kc * 32;
    v8i ai, biw;
#pragma unroll
    for (int v = 0; v < 8; ++v) {
      // A-layout: lane m=col, K = kbase + 16h + 2v
      const int ka = kbase + 16*hhalf + 2*v;
      const float a0 = (ka     < F_) ? feats[(size_t)(m0+col)*F_ + ka]     : 0.0f;
      const float a1 = (ka + 1 < F_) ? feats[(size_t)(m0+col)*F_ + ka + 1] : 0.0f;
      ai[v] = (int)pk_bf16(a0, a1);
      // B-layout: lane n=col+n0, K = kbase + 16*hhalf + 2v  (B[k][n] = W1[n][k])
      const int kb = kbase + 16*hhalf + 2*v;
      const int n  = n0 + col;
      const float w0 = (n < N1_ && kb     < F_) ? W1[(size_t)n*F_ + kb]     : 0.0f;
      const float w1 = (n < N1_ && kb + 1 < F_) ? W1[(size_t)n*F_ + kb + 1] : 0.0f;
      biw[v] = (int)pk_bf16(w0, w1);
    }
    v16bf Av = __builtin_bit_cast(v16bf, ai);
    v16bf Bv = __builtin_bit_cast(v16bf, biw);
    acc = __builtin_amdgcn_wmma_f32_16x16x32_bf16(false, Av, false, Bv,
                                                  (short)0, acc, false, false);
  }
  // D layout: lane: row m = r + 8*hhalf (+m0), col n = col (+n0)
#pragma unroll
  for (int r = 0; r < 8; ++r) {
    const int m = m0 + r + 8*hhalf;
    const int n = n0 + col;
    if (n < N1_) h1[(size_t)m*N1_ + n] = fmaxf(acc[r] + b1[n], 0.0f);
  }
}

// Layers 2+3 fused: tiny (64M MACs total) -> one batch row per lane, f32 VALU.
__global__ __launch_bounds__(256, 1)
void mlp23_kernel(const float* __restrict__ h1, const float* __restrict__ W2,
                  const float* __restrict__ b2, const float* __restrict__ W3,
                  const float* __restrict__ b3, float* __restrict__ out) {
  const int b = blockIdx.x * blockDim.x + threadIdx.x;
  if (b >= B_) return;
  const float* row = h1 + (size_t)b * N1_;
  float h2[N2_];
#pragma unroll 2
  for (int j = 0; j < N2_; ++j) {
    float acc = b2[j];
    const float* w = W2 + (size_t)j * N1_;
    for (int k = 0; k < N1_; ++k) acc = fmaf(row[k], w[k], acc);
    h2[j] = fmaxf(acc, 0.0f);
  }
#pragma unroll
  for (int i = 0; i < N3_; ++i) {
    float acc = b3[i];
#pragma unroll
    for (int j = 0; j < N2_; ++j) acc = fmaf(h2[j], W3[i*N2_ + j], acc);
    out[(size_t)b*N3_ + i] = fmaxf(acc, 0.0f);
  }
}

extern "C" void kernel_launch(void* const* d_in, const int* in_sizes, int n_in,
                              void* d_out, int out_size, void* d_ws, size_t ws_size,
                              hipStream_t stream) {
  (void)in_sizes; (void)n_in; (void)out_size; (void)ws_size;
  const float* x       = (const float*)d_in[0];
  const int*   lengths = (const int*)  d_in[1];
  const float* Wih     = (const float*)d_in[2];
  const float* Whh     = (const float*)d_in[3];
  const float* bih     = (const float*)d_in[4];
  const float* bhh     = (const float*)d_in[5];
  const float* W1      = (const float*)d_in[6];
  const float* b1      = (const float*)d_in[7];
  const float* W2      = (const float*)d_in[8];
  const float* b2      = (const float*)d_in[9];
  const float* W3      = (const float*)d_in[10];
  const float* b3      = (const float*)d_in[11];

  float* feats = (float*)d_ws;                  // 4096*156 f32 = 2.44 MB
  float* h1    = feats + (size_t)B_ * F_;       // 4096*300 f32 = 4.69 MB
  float* outp  = (float*)d_out;                 // 4096*14  f32

  rnn_kernel  <<<C_ * (B_ / 32), 64,  0, stream>>>(x, lengths, Wih, Whh, bih, bhh, feats);
  mlp1_kernel <<<(256 * 19) / 8, 256, 0, stream>>>(feats, W1, b1, h1);
  mlp23_kernel<<<B_ / 256,       256, 0, stream>>>(h1, W2, b2, W3, b3, outp);
}
